// RecurrentA3TGCN_40037685133526
// MI455X (gfx1250) — matrix operand order repaired and checked by reading, not compiled, not run on hardware
//
#include <hip/hip_runtime.h>
#include <hip/hip_bf16.h>

// ---------------------------------------------------------------------------
// A3TGCN on MI455X (gfx1250), compile-only target.
//
// N=50000 nodes, E=1.6M edges, HID=64, T=12 periods, IN=OUT=1.
//
// Key simplifications (exact, from IN==1):
//   conv_g(x_t)[n,:] = agg_t[n] * W_g + b_g,   agg_t[n] scalar
//   gate_pre = agg[n]*u_g + c_g + H @ Lg2      (u_g = W_g@Lg_top, 64x64 Lg2)
//   recurrence is per-node independent -> H, acc stay in VGPRs for all 12 steps
//
// Roofline: 14.7 GFLOP of 64-K GEMM (f16 WMMA -> negligible) vs ~120 MB HBM
// for the edge pass (~5 us @ 23.3 TB/s). agg (2.4 MB) and H-tile traffic stay
// in L2/LDS/registers. Kernel is edge-scatter bound by construction.
// ---------------------------------------------------------------------------

#define NN   50000
#define EE   1600000
#define TT   12
#define HIDN 64

typedef __attribute__((ext_vector_type(16))) _Float16 v16h;
typedef __attribute__((ext_vector_type(8)))  float    v8f;
typedef __attribute__((ext_vector_type(4)))  unsigned u32x4;
typedef __attribute__((ext_vector_type(8)))  int      i32x8;
typedef __attribute__((ext_vector_type(4)))  int      i32x4;

// ---------------------------------------------------------------- K0: deg init
__global__ void k_deg_init(float* __restrict__ deg) {
  int n = blockIdx.x * blockDim.x + threadIdx.x;
  if (n < NN) deg[n] = 1.0f;  // self-loop weight 1.0
}

// ------------------------------------------------------------ K1: deg scatter
__global__ void k_deg_acc(const int* __restrict__ ei, const float* __restrict__ ew,
                          float* __restrict__ deg) {
  int e = blockIdx.x * blockDim.x + threadIdx.x;
  if (e >= EE) return;
  atomicAdd(deg + ei[EE + e], ew[e]);
}

// --------------------------------- K2: dis = rsqrt(deg); agg init (self-loops)
__global__ void k_dis_self(const float* __restrict__ x, float* __restrict__ dis,
                           float* __restrict__ agg) {
  int n = blockIdx.x * blockDim.x + threadIdx.x;
  if (n >= NN) return;
  float d  = dis[n];                       // currently holds deg
  float di = (d > 0.0f) ? rsqrtf(d) : 0.0f;
  dis[n] = di;
  float sn = di * di;                      // self-loop norm = dis*1*dis
  const float* xr = x + (size_t)n * TT;
#pragma unroll
  for (int t = 0; t < TT; ++t) agg[t * NN + n] = xr[t] * sn;
}

// ----------------------- K3: fused edge pass: norm inline + 12-step scatter
__global__ void k_edge_scatter(const float* __restrict__ x, const int* __restrict__ ei,
                               const float* __restrict__ ew, const float* __restrict__ dis,
                               float* __restrict__ agg) {
  int e = blockIdx.x * blockDim.x + threadIdx.x;
  if (e >= EE) return;
  int s = ei[e], d = ei[EE + e];
  // prefetch a far-ahead x row (gfx1250 global_prefetch_b8)
  int e2 = e + 131072;
  if (e2 < EE) __builtin_prefetch(x + (size_t)ei[e2] * TT, 0, 0);
  float nw = dis[s] * ew[e] * dis[d];
  const float4* xr = (const float4*)(x + (size_t)s * TT);  // 48B rows, 16B aligned
  float4 x0 = xr[0], x1 = xr[1], x2 = xr[2];
  float xs[TT] = {x0.x, x0.y, x0.z, x0.w, x1.x, x1.y, x1.z, x1.w, x2.x, x2.y, x2.z, x2.w};
#pragma unroll
  for (int t = 0; t < TT; ++t) atomicAdd(agg + t * NN + d, xs[t] * nw);
}

// -------- K4: prep: softmax(att), fused u/c vectors, f16 B-fragment weights
// uvecs layout: [uz(64), cz(64), ur, cr, uh, ch]
// Bws layout:   [gate(z,r,h)][frag = nb*2+kc (8)][lane (32)][16 halves]
//   half i of lane L in frag (nb,kc) = L2[k][n], n = nb*16 + (L&15),
//   k = kc*32 + 16*(L>=16) + i      (contiguous k order, matches buildA)
__global__ void k_prep(const float* __restrict__ Wz, const float* __restrict__ bz,
                       const float* __restrict__ Wr, const float* __restrict__ br,
                       const float* __restrict__ Wh, const float* __restrict__ bh,
                       const float* __restrict__ Lz, const float* __restrict__ lbz,
                       const float* __restrict__ Lr, const float* __restrict__ lbr,
                       const float* __restrict__ Lh, const float* __restrict__ lbh,
                       const float* __restrict__ att,
                       float* __restrict__ uvecs, float* __restrict__ probs,
                       _Float16* __restrict__ Bws) {
  int tid = threadIdx.x;
  if (tid == 0) {  // softmax over 12 attention logits
    float mx = att[0];
    for (int t = 1; t < TT; ++t) mx = fmaxf(mx, att[t]);
    float ev[TT], s = 0.0f;
    for (int t = 0; t < TT; ++t) { ev[t] = __expf(att[t] - mx); s += ev[t]; }
    float inv = 1.0f / s;
    for (int t = 0; t < TT; ++t) probs[t] = ev[t] * inv;
  }
  for (int idx = tid; idx < 6 * HIDN; idx += blockDim.x) {
    int g  = idx >> 7;          // 0=z 1=r 2=h
    int uc = (idx >> 6) & 1;    // 0=u 1=c
    int j  = idx & 63;
    const float* W  = (g == 0) ? Wz : (g == 1) ? Wr : Wh;
    const float* B  = (g == 0) ? bz : (g == 1) ? br : bh;
    const float* L  = (g == 0) ? Lz : (g == 1) ? Lr : Lh;
    const float* lb = (g == 0) ? lbz : (g == 1) ? lbr : lbh;
    float s = 0.0f;
    if (uc == 0) {
      for (int k = 0; k < HIDN; ++k) s += W[k] * L[k * HIDN + j];
    } else {
      for (int k = 0; k < HIDN; ++k) s += B[k] * L[k * HIDN + j];
      s += lb[j];
    }
    uvecs[idx] = s;
  }
  for (int idx = tid; idx < 3 * 4096; idx += blockDim.x) {
    int g    = idx >> 12;
    int r    = idx & 4095;
    int frag = r >> 9;
    int lane = (r >> 4) & 31;
    int i    = r & 15;
    int nb = frag >> 1, kc = frag & 1;
    int n = nb * 16 + (lane & 15);
    int k = kc * 32 + ((lane >= 16) ? 16 : 0) + i;
    const float* L = (g == 0) ? Lz : (g == 1) ? Lr : Lh;
    Bws[idx] = (_Float16)L[(HIDN + k) * HIDN + n];  // bottom 64 rows of L
  }
}

// ------------------------------------------------- K5: fused 12-step GRU+attn
__device__ __forceinline__ v16h loadBfrag(const _Float16* sB, int frag, int lane) {
  return *(const v16h*)(sB + ((frag * 32 + lane) << 4));  // 32B aligned
}

// A operand (16-bit, 16x32): lane L holds row m=L&15; K halves contiguous:
// lanes 0-15: K 0..15; lanes 16-31: K 16..31 (matches B packing).
__device__ __forceinline__ void buildA(const _Float16* sHw, int col, int hi,
                                       v16h& a0, v16h& a1) {
#pragma unroll
  for (int i = 0; i < 16; ++i) {
    int kl = (hi << 4) + i;
    a0[i] = sHw[col * HIDN + kl];
    a1[i] = sHw[col * HIDN + 32 + kl];
  }
}

__global__ void __launch_bounds__(256)
k_gru_wmma(const float* __restrict__ agg, const float* __restrict__ uvecs,
           const float* __restrict__ probs, const _Float16* __restrict__ Bws,
           const float* __restrict__ Wout, const float* __restrict__ bout,
           float* __restrict__ out) {
  __shared__ __attribute__((aligned(32))) _Float16 sB[3 * 4096];     // 24 KB weights
  __shared__ __attribute__((aligned(32))) _Float16 sH[8][16 * HIDN]; // per-wave staging

  const int tid  = threadIdx.x;
  const int wave = tid >> 5;
  const int lane = tid & 31;
  const int hi   = lane >> 4;   // half-wave group (C-layout M offset 0/8)
  const int col  = lane & 15;   // C-layout column / A-layout row
  const int tileBase = blockIdx.x * 128 + wave * 16;

  // ---- stage gate matrices into LDS via Tensor Data Mover (wave 0) ----
#if defined(__HIP_DEVICE_COMPILE__) && __has_builtin(__builtin_amdgcn_tensor_load_to_lds)
  if (wave == 0) {
    unsigned lds = (unsigned)(unsigned long long)(const void*)&sB[0];
    unsigned long long ga = (unsigned long long)Bws;
    // D# group0: count=1, lds_addr, global_addr[56:0], type=2
    u32x4 g0 = {1u, lds, (unsigned)(ga & 0xFFFFFFFFull),
                (unsigned)((ga >> 32) & 0x01FFFFFFull) | 0x80000000u};
    // D# group1: data_size=3 (8B), tensor_dim0=tile_dim0=stride=3072, dim1=1
    i32x8 g1 = {0x00030000, (int)(3072u << 16), (int)(1u << 16),
                (int)(3072u << 16), 1, 3072, 0, 0};
    i32x4 gz = {0, 0, 0, 0};
#if __clang_major__ >= 23
    i32x8 gz8 = {0, 0, 0, 0, 0, 0, 0, 0};
    __builtin_amdgcn_tensor_load_to_lds(g0, g1, gz, gz, gz8, 0);
#else
    __builtin_amdgcn_tensor_load_to_lds(g0, g1, gz, gz, 0);
#endif
    __builtin_amdgcn_s_wait_tensorcnt(0);
  }
  __syncthreads();
#else
  for (int i = tid; i < 3 * 4096 / 2; i += 256)
    ((unsigned*)sB)[i] = ((const unsigned*)Bws)[i];
  __syncthreads();
#endif

  // ---- hoisted per-lane column constants ----
  float uzv[4], czv[4], urv[4], crv[4], uhv[4], chv[4], wov[4];
#pragma unroll
  for (int nb = 0; nb < 4; ++nb) {
    int j = nb * 16 + col;
    uzv[nb] = uvecs[0 * 64 + j];  czv[nb] = uvecs[1 * 64 + j];
    urv[nb] = uvecs[2 * 64 + j];  crv[nb] = uvecs[3 * 64 + j];
    uhv[nb] = uvecs[4 * 64 + j];  chv[nb] = uvecs[5 * 64 + j];
    wov[nb] = Wout[j];
  }

  // H and attention accumulator live in registers (C-layout: Vv <-> M=v+8*hi)
  v8f Hc[4], Ac[4];
#pragma unroll
  for (int nb = 0; nb < 4; ++nb)
#pragma unroll
    for (int v = 0; v < 8; ++v) { Hc[nb][v] = 0.0f; Ac[nb][v] = 0.0f; }

  _Float16* sHw = &sH[wave][0];

  for (int t = 0; t < TT; ++t) {
    // per-row scalar aggregates (clamped for the ragged last tile; EXEC stays full)
    float aggv[8];
#pragma unroll
    for (int v = 0; v < 8; ++v) {
      int node = tileBase + hi * 8 + v;
      if (node >= NN) node = NN - 1;
      aggv[v] = agg[t * NN + node];
    }

    // stage H (f32 -> f16, row-major 16x64) for A-fragment rebuild
#pragma unroll
    for (int nb = 0; nb < 4; ++nb)
#pragma unroll
      for (int v = 0; v < 8; ++v)
        sHw[(hi * 8 + v) * HIDN + nb * 16 + col] = (_Float16)Hc[nb][v];
    v16h aH0, aH1;
    buildA(sHw, col, hi, aH0, aH1);

    // Z gate: sigmoid(agg*uz + cz + H@Lz2)
    v8f Zc[4];
#pragma unroll
    for (int nb = 0; nb < 4; ++nb) {
      v8f c;
#pragma unroll
      for (int v = 0; v < 8; ++v) c[v] = aggv[v] * uzv[nb] + czv[nb];
      c = __builtin_amdgcn_wmma_f32_16x16x32_f16(false, aH0, false,
            loadBfrag(sB, 0 + nb * 2 + 0, lane), (short)0, c, false, false);
      c = __builtin_amdgcn_wmma_f32_16x16x32_f16(false, aH1, false,
            loadBfrag(sB, 0 + nb * 2 + 1, lane), (short)0, c, false, false);
#pragma unroll
      for (int v = 0; v < 8; ++v) c[v] = 1.0f / (1.0f + __expf(-c[v]));
      Zc[nb] = c;
    }

    // R gate
    v8f Rc[4];
#pragma unroll
    for (int nb = 0; nb < 4; ++nb) {
      v8f c;
#pragma unroll
      for (int v = 0; v < 8; ++v) c[v] = aggv[v] * urv[nb] + crv[nb];
      c = __builtin_amdgcn_wmma_f32_16x16x32_f16(false, aH0, false,
            loadBfrag(sB, 8 + nb * 2 + 0, lane), (short)0, c, false, false);
      c = __builtin_amdgcn_wmma_f32_16x16x32_f16(false, aH1, false,
            loadBfrag(sB, 8 + nb * 2 + 1, lane), (short)0, c, false, false);
#pragma unroll
      for (int v = 0; v < 8; ++v) c[v] = 1.0f / (1.0f + __expf(-c[v]));
      Rc[nb] = c;
    }

    // candidate: tanh(agg*uh + ch + (H*R)@Lh2) — restage H*R (wave-local LDS,
    // DS ops are in-order within a wave so the aH reads above are safe)
#pragma unroll
    for (int nb = 0; nb < 4; ++nb)
#pragma unroll
      for (int v = 0; v < 8; ++v)
        sHw[(hi * 8 + v) * HIDN + nb * 16 + col] = (_Float16)(Hc[nb][v] * Rc[nb][v]);
    v16h aR0, aR1;
    buildA(sHw, col, hi, aR0, aR1);

    float p = probs[t];
#pragma unroll
    for (int nb = 0; nb < 4; ++nb) {
      v8f c;
#pragma unroll
      for (int v = 0; v < 8; ++v) c[v] = aggv[v] * uhv[nb] + chv[nb];
      c = __builtin_amdgcn_wmma_f32_16x16x32_f16(false, aR0, false,
            loadBfrag(sB, 16 + nb * 2 + 0, lane), (short)0, c, false, false);
      c = __builtin_amdgcn_wmma_f32_16x16x32_f16(false, aR1, false,
            loadBfrag(sB, 16 + nb * 2 + 1, lane), (short)0, c, false, false);
#pragma unroll
      for (int v = 0; v < 8; ++v) {
        float ht = tanhf(c[v]);
        float z  = Zc[nb][v];
        float hn = z * Hc[nb][v] + (1.0f - z) * ht;
        Hc[nb][v] = hn;
        Ac[nb][v] += p * hn;
      }
    }
  }

  // ---- out[n] = relu(acc[n]) . Wout + bout (reduce 16 lanes per half-group) ----
  float bo = bout[0];
#pragma unroll
  for (int v = 0; v < 8; ++v) {
    float partial = 0.0f;
#pragma unroll
    for (int nb = 0; nb < 4; ++nb) {
      float a = Ac[nb][v];
      a = (a > 0.0f) ? a : 0.0f;
      partial += a * wov[nb];
    }
#pragma unroll
    for (int off = 1; off < 16; off <<= 1)
      partial += __shfl_xor(partial, off, 32);  // stays within 16-lane halves
    int node = tileBase + hi * 8 + v;
    if (col == 0 && node < NN) out[node] = partial + bo;
  }
}

// ---------------------------------------------------------------------------
extern "C" void kernel_launch(void* const* d_in, const int* in_sizes, int n_in,
                              void* d_out, int out_size, void* d_ws, size_t ws_size,
                              hipStream_t stream) {
  (void)in_sizes; (void)n_in; (void)out_size; (void)ws_size;
  const float* x    = (const float*)d_in[0];
  const int*   ei   = (const int*)d_in[1];
  const float* ew   = (const float*)d_in[2];
  const float* Wz   = (const float*)d_in[3];
  const float* bz   = (const float*)d_in[4];
  const float* Wr   = (const float*)d_in[5];
  const float* br   = (const float*)d_in[6];
  const float* Wh   = (const float*)d_in[7];
  const float* bh   = (const float*)d_in[8];
  const float* Lz   = (const float*)d_in[9];
  const float* lbz  = (const float*)d_in[10];
  const float* Lr   = (const float*)d_in[11];
  const float* lbr  = (const float*)d_in[12];
  const float* Lh   = (const float*)d_in[13];
  const float* lbh  = (const float*)d_in[14];
  const float* att  = (const float*)d_in[15];
  const float* Wout = (const float*)d_in[16];
  const float* bout = (const float*)d_in[17];
  float* out = (float*)d_out;

  // workspace carve-up (~2.6 MB): dis | agg[12][N] | uvecs | probs | f16 B weights
  float* fws   = (float*)d_ws;
  float* dis   = fws;                     // N
  float* agg   = fws + NN;                // TT*NN
  float* uvecs = agg + (size_t)TT * NN;   // 384
  float* probs = uvecs + 6 * HIDN;        // 12 (+pad)
  _Float16* Bws = (_Float16*)(probs + 16);  // 3*4096 halves, 32B aligned

  k_deg_init<<<(NN + 255) / 256, 256, 0, stream>>>(dis);
  k_deg_acc<<<(EE + 255) / 256, 256, 0, stream>>>(ei, ew, dis);
  k_dis_self<<<(NN + 255) / 256, 256, 0, stream>>>(x, dis, agg);
  k_edge_scatter<<<(EE + 255) / 256, 256, 0, stream>>>(x, ei, ew, dis, agg);
  k_prep<<<1, 256, 0, stream>>>(Wz, bz, Wr, br, Wh, bh, Lz, lbz, Lr, lbr, Lh, lbh,
                                att, uvecs, probs, Bws);
  k_gru_wmma<<<(NN + 127) / 128, 256, 0, stream>>>(agg, uvecs, probs, Bws, Wout, bout, out);
}
